// SelfAttention_73126113182000
// MI455X (gfx1250) — matrix operand order repaired and checked by reading, not compiled, not run on hardware
//
#include <hip/hip_runtime.h>

// ---------------------------------------------------------------------------
// Fused self-attention for MI455X (gfx1250, wave32, WMMA, async-to-LDS, TDM).
//   q,k,v = x @ W^T + b        (bf16 WMMA 16x16x32, fp32 accumulate)
//   out   = softmax(q k^T) v   (flash-style; K chunks double-buffered into LDS
//                               via GLOBAL_LOAD_ASYNC_TO_LDS, Q tile via TDM)
// ---------------------------------------------------------------------------

#define DI __device__ __forceinline__
#define GLOBAL_AS __attribute__((address_space(1)))
#define LDS_AS    __attribute__((address_space(3)))

typedef __attribute__((ext_vector_type(4)))  __bf16 v4bf;
typedef __attribute__((ext_vector_type(8)))  __bf16 v8bf;
typedef __attribute__((ext_vector_type(16))) __bf16 v16bf;
typedef __attribute__((ext_vector_type(8)))  float  v8f;
typedef __attribute__((ext_vector_type(4)))  unsigned u32x4;
typedef __attribute__((ext_vector_type(4)))  int      i32x4;
typedef __attribute__((ext_vector_type(8)))  int      i32x8;
typedef int b128_t __attribute__((vector_size(16)));   // async builtin pointee

constexpr int B = 8, S = 2048, E = 1024;
constexpr int KPAD = 8;   // bf16 elems of row pad in LDS K tiles (bank rotate)
constexpr int FPAD = 4;   // fp32 elems of row pad in LDS x tile

#if defined(__has_builtin)
#  if __has_builtin(__builtin_amdgcn_global_load_async_to_lds_b128)
#    define HAS_ASYNC_B 1
#  endif
#  if __has_builtin(__builtin_amdgcn_tensor_load_to_lds)
#    define HAS_TDM 1
#  endif
#endif

// ---- CDNA5 async copy: one 16-byte segment global -> LDS (ASYNCcnt) -------
DI void async_copy_b128(const void* g, void* l) {
#if defined(HAS_ASYNC_B)
  __builtin_amdgcn_global_load_async_to_lds_b128(
      (GLOBAL_AS b128_t*)g, (LDS_AS b128_t*)l, 0, 0);
#else
  asm volatile("global_load_async_to_lds_b128 %0, %1, off"
               :: "v"((LDS_AS void*)l), "v"(g) : "memory");
#endif
}

DI void wait_async0() {
#if defined(__has_builtin) && __has_builtin(__builtin_amdgcn_s_wait_asynccnt)
  __builtin_amdgcn_s_wait_asynccnt(0);
#else
  asm volatile("s_wait_asynccnt 0x0" ::: "memory");
#endif
}

DI void wait_tensor0() {
#if defined(__has_builtin) && __has_builtin(__builtin_amdgcn_s_wait_tensorcnt)
  __builtin_amdgcn_s_wait_tensorcnt(0);
#else
  asm volatile("s_wait_tensorcnt 0x0" ::: "memory");
#endif
}

#if defined(HAS_TDM)
// ---- TDM: 2D tile (rows x rowElems bf16) global -> LDS, one descriptor ----
DI void tdm_load_2d_bf16(const void* gsrc, void* ldst,
                         unsigned rows, unsigned rowElems,
                         unsigned rowStrideElems) {
  unsigned long long ga = (unsigned long long)gsrc;
  unsigned ldsOff = (unsigned)(__UINTPTR_TYPE__)(LDS_AS void*)ldst;
  u32x4 g0;
  g0[0] = 1u;                                        // count=1, user descriptor
  g0[1] = ldsOff;                                    // lds_addr
  g0[2] = (unsigned)ga;                              // global_addr[31:0]
  g0[3] = (unsigned)((ga >> 32) & 0x01FFFFFFu) | 0x80000000u;  // addr[56:32]|type=2
  i32x8 g1;
  g1[0] = 0x00010000;                                // data_size = 2 bytes
  g1[1] = (int)(rowElems << 16);                     // tensor_dim0[15:0]
  g1[2] = (int)((rowElems >> 16) | (rows << 16));    // dim0 hi | tensor_dim1 lo
  g1[3] = (int)((rows >> 16) | (rowElems << 16));    // dim1 hi | tile_dim0
  g1[4] = (int)rows;                                 // tile_dim1 (tile_dim2=0)
  g1[5] = (int)rowStrideElems;                       // tensor_dim0_stride lo32
  g1[6] = 0;
  g1[7] = 0;                                         // dim1_stride unused (2D)
  i32x4 z4; i32x8 z8;
#pragma unroll
  for (int i = 0; i < 4; ++i) z4[i] = 0;
#pragma unroll
  for (int i = 0; i < 8; ++i) z8[i] = 0;
#if __clang_major__ >= 23
  __builtin_amdgcn_tensor_load_to_lds(g0, g1, z4, z4, z8, 0);
#else
  __builtin_amdgcn_tensor_load_to_lds(g0, g1, z4, z4, 0);
#endif
}
#endif

DI v16bf cat16(v8bf lo, v8bf hi) {
  v16bf r;
#pragma unroll
  for (int i = 0; i < 8; ++i) { r[i] = lo[i]; r[i + 8] = hi[i]; }
  return r;
}

DI v8f vzero8() {
  v8f z;
#pragma unroll
  for (int i = 0; i < 8; ++i) z[i] = 0.f;
  return z;
}

DI v8f wmma_bf16(v16bf a, v16bf b, v8f c) {
  return __builtin_amdgcn_wmma_f32_16x16x32_bf16(
      false, a, false, b, (short)0, c, false, false);
}

// ---------------------------------------------------------------------------
// Kernel 1: Wq/Wk/Wv fp32 -> bf16
// ---------------------------------------------------------------------------
__global__ __launch_bounds__(256)
void cvt_weights(const float* __restrict__ Wq, const float* __restrict__ Wk,
                 const float* __restrict__ Wv, __bf16* __restrict__ wOut) {
  const int per4 = (E * E) / 4;
  int idx4 = blockIdx.x * 256 + threadIdx.x;
  int mat  = idx4 / per4;
  int off4 = idx4 - mat * per4;
  const float* src = (mat == 0) ? Wq : (mat == 1 ? Wk : Wv);
  float4 v = ((const float4*)src)[off4];
  v4bf o;
  o[0] = (__bf16)v.x; o[1] = (__bf16)v.y; o[2] = (__bf16)v.z; o[3] = (__bf16)v.w;
  *(v4bf*)(wOut + (size_t)mat * E * E + (size_t)off4 * 4) = o;
}

// ---------------------------------------------------------------------------
// Kernel 2: QKV projection. x tile staged fp32 -> LDS with async-to-LDS.
// ---------------------------------------------------------------------------
__global__ __launch_bounds__(256)
void qkv_proj(const float* __restrict__ x, const __bf16* __restrict__ wAll,
              const float* __restrict__ bq, const float* __restrict__ bk,
              const float* __restrict__ bv,
              __bf16* __restrict__ qOut, __bf16* __restrict__ kOut,
              __bf16* __restrict__ vTOut) {
  __shared__ float sXf[16 * (E + FPAD)];              // ~66 KB fp32, padded

  const int m0   = blockIdx.x * 16;
  const int tid  = threadIdx.x;
  const int lane = tid & 31;
  const int wave = tid >> 5;
  const int n15  = lane & 15;
  const int lg   = lane >> 4;

  // Async-stage the 16xE fp32 x tile: 4096 16B segments, 16 per thread.
#pragma unroll
  for (int j = 0; j < 16; ++j) {
    const int s   = tid + 256 * j;
    const int row = s >> 8;                           // 256 segments per row
    const int c4  = s & 255;                          // float4 index in row
    async_copy_b128(x + (size_t)(m0 + row) * E + c4 * 4,
                    &sXf[row * (E + FPAD) + c4 * 4]);
  }
  wait_async0();
  __syncthreads();

  const int f0    = wave * 128;
  const int bIdx  = m0 / S;
  const int sLoc0 = m0 - bIdx * S;
  const float* biasPtr[3] = {bq, bk, bv};

#pragma unroll
  for (int mat = 0; mat < 3; ++mat) {
    const __bf16* w = wAll + (size_t)mat * E * E;
    v8f acc[8];
#pragma unroll
    for (int nt = 0; nt < 8; ++nt) acc[nt] = vzero8();

    for (int e0 = 0; e0 < E; e0 += 32) {
      // Build bf16 A tile (16x32) from fp32 LDS.
      const float* ab = &sXf[n15 * (E + FPAD) + e0];
      float4 f0a = *(const float4*)(ab + 8 * lg);
      float4 f0b = *(const float4*)(ab + 8 * lg + 4);
      float4 f1a = *(const float4*)(ab + 16 + 8 * lg);
      float4 f1b = *(const float4*)(ab + 16 + 8 * lg + 4);
      v16bf a;
      a[0]  = (__bf16)f0a.x; a[1]  = (__bf16)f0a.y;
      a[2]  = (__bf16)f0a.z; a[3]  = (__bf16)f0a.w;
      a[4]  = (__bf16)f0b.x; a[5]  = (__bf16)f0b.y;
      a[6]  = (__bf16)f0b.z; a[7]  = (__bf16)f0b.w;
      a[8]  = (__bf16)f1a.x; a[9]  = (__bf16)f1a.y;
      a[10] = (__bf16)f1a.z; a[11] = (__bf16)f1a.w;
      a[12] = (__bf16)f1b.x; a[13] = (__bf16)f1b.y;
      a[14] = (__bf16)f1b.z; a[15] = (__bf16)f1b.w;
#pragma unroll
      for (int nt = 0; nt < 8; ++nt) {
        const __bf16* wp = w + (size_t)(f0 + nt * 16 + n15) * E + e0 + 8 * lg;
        v16bf bm = cat16(*(const v8bf*)wp, *(const v8bf*)(wp + 16));
        acc[nt] = wmma_bf16(a, bm, acc[nt]);
      }
    }

    const float* bp = biasPtr[mat];
#pragma unroll
    for (int nt = 0; nt < 8; ++nt) {
      const int col   = f0 + nt * 16 + n15;
      const float bv_ = bp[col];
#pragma unroll
      for (int r = 0; r < 8; ++r) {
        const int row  = r + 8 * lg;
        const float vv = acc[nt][r] + bv_;
        if (mat == 0)
          qOut[(size_t)(m0 + row) * E + col] = (__bf16)vv;
        else if (mat == 1)
          kOut[(size_t)(m0 + row) * E + col] = (__bf16)vv;
        else
          vTOut[((size_t)bIdx * E + col) * S + sLoc0 + row] = (__bf16)vv;
      }
    }
  }
}

// ---------------------------------------------------------------------------
// Kernel 3: flash attention.
//   Q tile: one TDM tensor_load_to_lds per block (TENSORcnt).
//   K chunks: double-buffered GLOBAL_LOAD_ASYNC_TO_LDS (ASYNCcnt), copy of
//   chunk i+1 overlaps all compute of chunk i.
// ---------------------------------------------------------------------------
constexpr int SROW = 36;                              // padded score row (fp32)
constexpr int KROW = E + KPAD;                        // padded K row (bf16)

DI void stage_k_async(const __bf16* kb, int kc, __bf16* dst, int tid) {
  // 32 rows x 1024 bf16 = 4096 16B segments; 16 per thread, coalesced.
#pragma unroll
  for (int j = 0; j < 16; ++j) {
    const int s   = tid + 256 * j;
    const int row = s >> 7;                           // 128 segments per row
    const int c8  = s & 127;                          // 8-elem segment in row
    async_copy_b128(kb + (size_t)(kc + row) * E + c8 * 8,
                    dst + row * KROW + c8 * 8);
  }
}

__global__ __launch_bounds__(256)
void flash_attn(const __bf16* __restrict__ qAll, const __bf16* __restrict__ kAll,
                const __bf16* __restrict__ vTAll, float* __restrict__ out) {
  __shared__ __bf16 sK[2][32 * KROW];                 // 129 KB double buffer
  __shared__ __bf16 sQ[16 * E];                       // 32 KB (TDM destination)
  __shared__ float  sS[8 * 16 * SROW];                // 18 KB per-wave slots
  __shared__ __bf16 sP[16 * 32];
  __shared__ float  sAlpha[16];
  __shared__ float  sLinv[16];

  const int tid  = threadIdx.x;
  const int lane = tid & 31;
  const int wave = tid >> 5;
  const int n15  = lane & 15;
  const int lg   = lane >> 4;
  const int b    = blockIdx.y;
  const int q0   = blockIdx.x * 16;
  const int eB   = wave * 128;

  const __bf16* qb = qAll + ((size_t)b * S + q0) * E;
  const __bf16* kb = kAll + (size_t)b * S * E;
  const __bf16* vt = vTAll + (size_t)b * E * S;

  // Kick off chunk-0 K copy immediately.
  stage_k_async(kb, 0, &sK[0][0], tid);

#if defined(HAS_TDM)
  if (wave == 0) {                                    // one DMA for Q tile
    tdm_load_2d_bf16(qb, &sQ[0], /*rows=*/16, /*rowElems=*/E,
                     /*rowStrideElems=*/E);
    wait_tensor0();
  }
  __syncthreads();
  v16bf qA[4];
#pragma unroll
  for (int t = 0; t < 4; ++t) {
    const __bf16* p = &sQ[n15 * E + eB + 32 * t + 8 * lg];
    qA[t] = cat16(*(const v8bf*)p, *(const v8bf*)(p + 16));
  }
#else
  v16bf qA[4];
#pragma unroll
  for (int t = 0; t < 4; ++t) {
    const __bf16* p = qb + (size_t)n15 * E + eB + 32 * t + 8 * lg;
    qA[t] = cat16(*(const v8bf*)p, *(const v8bf*)(p + 16));
  }
#endif

  v8f o[8];
#pragma unroll
  for (int et = 0; et < 8; ++et) o[et] = vzero8();
  float mI = -1e30f, lI = 0.f;

  int bufSel = 0;
  for (int kc = 0; kc < S; kc += 32, bufSel ^= 1) {
    wait_async0();                                    // chunk kc resident
    __syncthreads();                                  // visible to all waves
    const __bf16* sKb = &sK[bufSel][0];
    if (kc + 32 < S)                                  // overlap next copy
      stage_k_async(kb, kc + 32, &sK[bufSel ^ 1][0], tid);

    // -------- partial scores from LDS K chunk ------------------------------
#pragma unroll
    for (int kt = 0; kt < 2; ++kt) {
      v8f c = vzero8();
#pragma unroll
      for (int t = 0; t < 4; ++t) {
        const __bf16* p = &sKb[(kt * 16 + n15) * KROW + eB + 32 * t + 8 * lg];
        c = wmma_bf16(qA[t], cat16(*(const v8bf*)p, *(const v8bf*)(p + 16)), c);
      }
#pragma unroll
      for (int r = 0; r < 8; ++r)
        sS[wave * (16 * SROW) + (r + 8 * lg) * SROW + n15 + 16 * kt] = c[r];
    }
    __syncthreads();

    // -------- online softmax (lane: row n15, cols [16lg, 16lg+16)) ---------
    float vals[16];
#pragma unroll
    for (int j = 0; j < 16; ++j) vals[j] = 0.f;
#pragma unroll
    for (int w = 0; w < 8; ++w) {                     // fixed order: determinism
      const float4* p = (const float4*)&sS[w * (16 * SROW) + n15 * SROW + 16 * lg];
#pragma unroll
      for (int j = 0; j < 4; ++j) {
        float4 t = p[j];
        vals[4 * j + 0] += t.x; vals[4 * j + 1] += t.y;
        vals[4 * j + 2] += t.z; vals[4 * j + 3] += t.w;
      }
    }
    float mx = vals[0];
#pragma unroll
    for (int j = 1; j < 16; ++j) mx = fmaxf(mx, vals[j]);
    mx = fmaxf(mx, __shfl_xor(mx, 16));
    const float mNew = fmaxf(mI, mx);
    float pv[16], sum = 0.f;
#pragma unroll
    for (int j = 0; j < 16; ++j) { pv[j] = __expf(vals[j] - mNew); sum += pv[j]; }
    sum += __shfl_xor(sum, 16);
    const float alpha = __expf(mI - mNew);
    lI = lI * alpha + sum;
    mI = mNew;

    if (wave == 0) {
      sAlpha[n15] = alpha;
      v8bf p0, p1;
#pragma unroll
      for (int j = 0; j < 8; ++j) { p0[j] = (__bf16)pv[j]; p1[j] = (__bf16)pv[j + 8]; }
      *(v8bf*)&sP[n15 * 32 + 16 * lg]     = p0;
      *(v8bf*)&sP[n15 * 32 + 16 * lg + 8] = p1;
    }
    __syncthreads();

    // -------- rescale O and accumulate P x V (V direct from L2) ------------
    float al[8];
#pragma unroll
    for (int r = 0; r < 8; ++r) al[r] = sAlpha[r + 8 * lg];
    const __bf16* pp = &sP[n15 * 32];
    v16bf pA = cat16(*(const v8bf*)(pp + 8 * lg),
                     *(const v8bf*)(pp + 16 + 8 * lg));
#pragma unroll
    for (int et = 0; et < 8; ++et) {
#pragma unroll
      for (int r = 0; r < 8; ++r) o[et][r] *= al[r];
      const __bf16* vp = vt + (size_t)(eB + et * 16 + n15) * S + kc + 8 * lg;
      o[et] = wmma_bf16(pA, cat16(*(const v8bf*)vp, *(const v8bf*)(vp + 16)), o[et]);
    }
    if (kc + 32 < S)
      __builtin_prefetch(vt + (size_t)(eB + n15) * S + kc + 32, 0, 0);
  }

  if (wave == 0) sLinv[n15] = 1.f / lI;
  __syncthreads();
  float li[8];
#pragma unroll
  for (int r = 0; r < 8; ++r) li[r] = sLinv[r + 8 * lg];
  float* ob = out + ((size_t)b * S + q0) * E + eB;
#pragma unroll
  for (int et = 0; et < 8; ++et)
#pragma unroll
    for (int r = 0; r < 8; ++r)
      ob[(size_t)(r + 8 * lg) * E + et * 16 + n15] = o[et][r] * li[r];
}

// ---------------------------------------------------------------------------
// Host launch. ws layout (bf16 elems): Wq|Wk|Wv (3*E*E), q, k (B*S*E each,
// row-major), vT (B*E*S). Total ~107 MB.
// ---------------------------------------------------------------------------
extern "C" void kernel_launch(void* const* d_in, const int* in_sizes, int n_in,
                              void* d_out, int out_size, void* d_ws, size_t ws_size,
                              hipStream_t stream) {
  (void)in_sizes; (void)n_in; (void)out_size; (void)ws_size;
  const float* x  = (const float*)d_in[0];
  const float* Wq = (const float*)d_in[1];
  const float* bq = (const float*)d_in[2];
  const float* Wk = (const float*)d_in[3];
  const float* bk = (const float*)d_in[4];
  const float* Wv = (const float*)d_in[5];
  const float* bv = (const float*)d_in[6];
  float* out = (float*)d_out;

  __bf16* ws   = (__bf16*)d_ws;
  const size_t EE  = (size_t)E * E;
  const size_t BSE = (size_t)B * S * E;
  __bf16* wAll = ws;
  __bf16* qb   = ws + 3 * EE;
  __bf16* kbuf = qb + BSE;
  __bf16* vT   = kbuf + BSE;

  cvt_weights<<<(3 * E * E / 4) / 256, 256, 0, stream>>>(Wq, Wk, Wv, wAll);
  qkv_proj<<<(B * S) / 16, 256, 0, stream>>>(x, wAll, bq, bk, bv, qb, kbuf, vT);
  flash_attn<<<dim3(S / 16, B), 256, 0, stream>>>(qb, kbuf, vT, out);
}
